// GatedRecurrentUnitHnNet_79207786872897
// MI455X (gfx1250) — matrix operand order repaired and checked by reading, not compiled, not run on hardware
//
#include <hip/hip_runtime.h>
#include <hip/hip_bf16.h>

// ---------------------------------------------------------------------------
// Types for CDNA5 WMMA (wave32, v_wmma_f32_16x16x32_bf16)
// ---------------------------------------------------------------------------
typedef __bf16 bf16_t;
typedef __attribute__((ext_vector_type(16))) __bf16 v16bf;
typedef __attribute__((ext_vector_type(8)))  __bf16 v8bf;
typedef __attribute__((ext_vector_type(8)))  float  v8f;
typedef __attribute__((ext_vector_type(4)))  int    v4i;

#define GRU_B   64
#define GRU_T   256
#define GRU_IN  128
#define GRU_H   512
#define GRU_3H  1536
#define GRU_M   (GRU_B * GRU_T)   // 16384 rows for input-side GEMMs
#define STEP_WGS 8
#define STEP_LDS (192 * GRU_H * 2)  // 196608 B: 3 gates x 64 cols x 512 K bf16

// ---------------------------------------------------------------------------
// Fragment loaders (ISA 7.12.2 layouts)
// A 16x32 bf16: lanes 0-15 hold K={kb..kb+7, kb+16..kb+23}, lanes 16-31 the
// other two 8-chunks -> two 16B loads per lane, interleaved.
// ---------------------------------------------------------------------------
__device__ __forceinline__ v16bf load_frag_a(const bf16_t* __restrict__ row,
                                             int kb, int sel) {
  const bf16_t* p = row + kb + sel * 8;
  v8bf lo = *(const v8bf*)(p);
  v8bf hi = *(const v8bf*)(p + 16);
  return __builtin_shufflevector(lo, hi, 0, 1, 2, 3, 4, 5, 6, 7,
                                 8, 9, 10, 11, 12, 13, 14, 15);
}

// B 32x16 bf16 (dense, column n = W row n): same K order as A -> lanes 0-15
// hold K={kb..kb+7, kb+16..kb+23}, lanes 16-31 the other two 8-chunks.
__device__ __forceinline__ v16bf load_frag_b(const bf16_t* __restrict__ row,
                                             int kb, int sel) {
  const bf16_t* p = row + kb + sel * 8;
  v8bf lo = *(const v8bf*)(p);
  v8bf hi = *(const v8bf*)(p + 16);
  return __builtin_shufflevector(lo, hi, 0, 1, 2, 3, 4, 5, 6, 7,
                                 8, 9, 10, 11, 12, 13, 14, 15);
}

__device__ __forceinline__ float sigm_f(float x) {
  float e = __expf(-fabsf(x));
  float s = 1.0f / (1.0f + e);
  return x >= 0.0f ? s : 1.0f - s;
}
__device__ __forceinline__ float tanh_f(float x) {
  float e = __expf(-2.0f * fabsf(x));
  float t = (1.0f - e) / (1.0f + e);
  return x >= 0.0f ? t : -t;
}

// ---------------------------------------------------------------------------
// f32 -> bf16 conversion
// ---------------------------------------------------------------------------
__global__ __launch_bounds__(256) void cvt_f32_bf16(const float* __restrict__ in,
                                                    bf16_t* __restrict__ out,
                                                    int n) {
  int i = blockIdx.x * 256 + threadIdx.x;
  if (i < n) out[i] = (bf16_t)in[i];
}

// ---------------------------------------------------------------------------
// xg = A[M,K](bf16) @ W[1536,K](bf16)^T + bias   -> C[M,1536] f32
// Block: 256 thr = 8 waves, tile 128(M) x 64(N); wave = 2x2 subtiles of 16.
// ---------------------------------------------------------------------------
__global__ __launch_bounds__(256) void gemm_bf16_bias(
    const bf16_t* __restrict__ A, const bf16_t* __restrict__ W,
    const float* __restrict__ bias, float* __restrict__ C,
    int M, int K) {
  const int tid  = threadIdx.x;
  const int lane = tid & 31;
  const int wave = tid >> 5;
  const int sel  = lane >> 4;
  const int ln   = lane & 15;
  const int mbase = blockIdx.x * 128 + (wave >> 1) * 32;
  const int nbase = blockIdx.y * 64 + (wave & 1) * 32;
  (void)M;

  v8f acc[2][2] = {};
  for (int kb = 0; kb < K; kb += 32) {
    // non-blocking prefetch of the next K tile (global_prefetch_b8)
    if (kb + 32 < K) {
      __builtin_prefetch(A + (size_t)(mbase + ln) * K + kb + 32, 0, 1);
      __builtin_prefetch(W + (size_t)(nbase + ln) * K + kb + 32, 0, 1);
    }
    v16bf a[2], b[2];
#pragma unroll
    for (int i = 0; i < 2; ++i)
      a[i] = load_frag_a(A + (size_t)(mbase + i * 16 + ln) * K, kb, sel);
#pragma unroll
    for (int j = 0; j < 2; ++j)
      b[j] = load_frag_b(W + (size_t)(nbase + j * 16 + ln) * K, kb, sel);
#pragma unroll
    for (int i = 0; i < 2; ++i)
#pragma unroll
      for (int j = 0; j < 2; ++j)
        acc[i][j] = __builtin_amdgcn_wmma_f32_16x16x32_bf16(
            false, a[i], false, b[j], (short)0, acc[i][j], false, false);
  }
#pragma unroll
  for (int j = 0; j < 2; ++j) {
    const int col = nbase + j * 16 + ln;
    const float bv = bias[col];
#pragma unroll
    for (int i = 0; i < 2; ++i) {
      const int row0 = mbase + i * 16 + (sel ? 8 : 0);
#pragma unroll
      for (int e = 0; e < 8; ++e)
        C[(size_t)(row0 + e) * GRU_3H + col] = acc[i][j][e] + bv;
    }
  }
}

// ---------------------------------------------------------------------------
// Persistent recurrent kernel: 8 WGs x 256 thr. WG g owns H-cols [g*64,g*64+64).
// Stages its 3x64x512 bf16 W_hh slice into 196KB LDS once (async-to-LDS when
// the toolchain exposes it), then iterates all T steps with an agent-scope
// atomic grid barrier. h double-buffered to avoid read/write races.
// ---------------------------------------------------------------------------
extern __shared__ bf16_t s_whh[];

__global__ __launch_bounds__(256) void gru_layer_kernel(
    const bf16_t* __restrict__ Whh,   // [1536,512] bf16 row-major
    const float*  __restrict__ bhh,   // [1536]
    const float*  __restrict__ xg,    // [B,T,1536] f32 (b_ih already folded)
    float* hF0, float* hF1,           // h f32 double buffer [64,512]
    bf16_t* hB0, bf16_t* hB1,         // h bf16 double buffer [64,512]
    bf16_t* __restrict__ outB,        // [B,T,512] bf16 hidden sequence
    int* __restrict__ bar) {
  const int tid = threadIdx.x;
  const int wg  = blockIdx.x;

  // ---- stage W_hh slice into LDS: local row r = gate*64 + c, 512 K each ----
#if __has_builtin(__builtin_amdgcn_global_load_async_to_lds_b128)
  {
    typedef __attribute__((address_space(1))) v4i* as1_v4ip;
    typedef __attribute__((address_space(3))) v4i* as3_v4ip;
    char* lds_base = (char*)s_whh;
    for (int i = tid; i < (192 * GRU_H * 2) / 16; i += 256) {
      const int r  = i >> 6;       // 64 x 16B chunks per 512-elem row
      const int kc = i & 63;
      const int g  = r >> 6;
      const int c  = r & 63;
      const char* src =
          (const char*)(Whh + (size_t)(g * GRU_H + wg * 64 + c) * GRU_H) +
          (size_t)kc * 16;
      __builtin_amdgcn_global_load_async_to_lds_b128(
          (as1_v4ip)src, (as3_v4ip)(lds_base + (size_t)i * 16), 0, 0);
    }
#if __has_builtin(__builtin_amdgcn_s_wait_asynccnt)
    __builtin_amdgcn_s_wait_asynccnt(0);
#else
    asm volatile("s_wait_asynccnt 0x0" ::: "memory");
#endif
  }
#else
  {
    uint4* dst = (uint4*)s_whh;
    for (int i = tid; i < (192 * GRU_H * 2) / 16; i += 256) {
      const int r  = i >> 6;
      const int kc = i & 63;
      const int g  = r >> 6;
      const int c  = r & 63;
      const uint4* src =
          (const uint4*)(Whh + (size_t)(g * GRU_H + wg * 64 + c) * GRU_H) + kc;
      dst[i] = *src;
    }
  }
#endif
  __syncthreads();

  const int lane = tid & 31;
  const int wave = tid >> 5;
  const int sel  = lane >> 4;
  const int ln   = lane & 15;
  const int mbase = (wave >> 1) * 16;     // batch-row tile (M=64 -> 4 tiles)
  const int arow  = mbase + ln;           // A-matrix row for this lane
  const int nloc0 = (wave & 1) * 32;      // local col base within WG slice

  // biases depend only on lane/subtile -> hoist out of the time loop
  float bh[3][2];
  int   colg[2];
#pragma unroll
  for (int s2 = 0; s2 < 2; ++s2) {
    colg[s2] = wg * 64 + nloc0 + s2 * 16 + ln;
#pragma unroll
    for (int g = 0; g < 3; ++g) bh[g][s2] = bhh[g * GRU_H + colg[s2]];
  }

  float*  hFbuf[2] = {hF0, hF1};
  bf16_t* hBbuf[2] = {hB0, hB1};

  for (int t = 0; t < GRU_T; ++t) {
    const int in  = t & 1;
    const int out = in ^ 1;
    const float*  hFin  = hFbuf[in];
    float*        hFout = hFbuf[out];
    const bf16_t* hBin  = hBbuf[in];
    bf16_t*       hBout = hBbuf[out];

    v8f acc[3][2] = {};
    for (int kb = 0; kb < GRU_H; kb += 32) {
      const v16bf a = load_frag_a(hBin + (size_t)arow * GRU_H, kb, sel);
#pragma unroll
      for (int g = 0; g < 3; ++g)
#pragma unroll
        for (int s2 = 0; s2 < 2; ++s2) {
          const v16bf b = load_frag_b(
              s_whh + (size_t)(g * 64 + nloc0 + s2 * 16 + ln) * GRU_H, kb, sel);
          acc[g][s2] = __builtin_amdgcn_wmma_f32_16x16x32_bf16(
              false, a, false, b, (short)0, acc[g][s2], false, false);
        }
    }

    // fused gate epilogue: element e -> row mbase+e+8*sel, col colg[s2]
#pragma unroll
    for (int s2 = 0; s2 < 2; ++s2) {
      const int cg = colg[s2];
#pragma unroll
      for (int e = 0; e < 8; ++e) {
        const int rr = mbase + e + (sel ? 8 : 0);     // batch index
        const size_t xoff = ((size_t)rr * GRU_T + t) * GRU_3H + cg;
        const float hr = acc[0][s2][e] + bh[0][s2];
        const float hz = acc[1][s2][e] + bh[1][s2];
        const float hn = acc[2][s2][e] + bh[2][s2];
        const float r  = sigm_f(xg[xoff] + hr);
        const float z  = sigm_f(xg[xoff + GRU_H] + hz);
        const float n  = tanh_f(xg[xoff + 2 * GRU_H] + r * hn);
        const size_t hoff = (size_t)rr * GRU_H + cg;
        const float hv = (1.0f - z) * n + z * hFin[hoff];
        hFout[hoff] = hv;
        const bf16_t hb = (bf16_t)hv;
        hBout[hoff] = hb;
        outB[((size_t)rr * GRU_T + t) * GRU_H + cg] = hb;
      }
    }

    // ---- grid barrier across the 8 persistent WGs ----
    __threadfence();
    __syncthreads();
    if (tid == 0)
      __hip_atomic_fetch_add(bar, 1, __ATOMIC_ACQ_REL, __HIP_MEMORY_SCOPE_AGENT);
    const int target = STEP_WGS * (t + 1);
    while (__hip_atomic_load(bar, __ATOMIC_ACQUIRE, __HIP_MEMORY_SCOPE_AGENT) <
           target)
      __builtin_amdgcn_s_sleep(1);
    __syncthreads();
  }
}

// ---------------------------------------------------------------------------
// out[b,o] = b_out[o] + h0[b,:]. Wout[o,0:512] + h1[b,:] . Wout[o,512:1024]
// ---------------------------------------------------------------------------
__global__ __launch_bounds__(256) void out_proj(
    const float* __restrict__ h0, const float* __restrict__ h1,
    const float* __restrict__ Wout, const float* __restrict__ bout,
    float* __restrict__ out) {
  const int g = blockIdx.x * 256 + threadIdx.x;  // 4096 = 64 x 64
  const int b = g >> 6;
  const int o = g & 63;
  float s = bout[o];
  const float* w = Wout + (size_t)o * (2 * GRU_H);
  for (int j = 0; j < GRU_H; ++j) s += h0[(size_t)b * GRU_H + j] * w[j];
  for (int j = 0; j < GRU_H; ++j) s += h1[(size_t)b * GRU_H + j] * w[GRU_H + j];
  out[(size_t)b * 64 + o] = s;
}

// ---------------------------------------------------------------------------
// Host side
// ---------------------------------------------------------------------------
extern "C" void kernel_launch(void* const* d_in, const int* in_sizes, int n_in,
                              void* d_out, int out_size, void* d_ws,
                              size_t ws_size, hipStream_t stream) {
  (void)in_sizes; (void)n_in; (void)out_size; (void)ws_size;
  const float* x    = (const float*)d_in[0];
  const float* Wih0 = (const float*)d_in[1];
  const float* Whh0 = (const float*)d_in[2];
  const float* bih0 = (const float*)d_in[3];
  const float* bhh0 = (const float*)d_in[4];
  const float* Wih1 = (const float*)d_in[5];
  const float* Whh1 = (const float*)d_in[6];
  const float* bih1 = (const float*)d_in[7];
  const float* bhh1 = (const float*)d_in[8];
  const float* Wout = (const float*)d_in[9];
  const float* bout = (const float*)d_in[10];
  float* out = (float*)d_out;

  // workspace carve-up (256B aligned)
  char* w = (char*)d_ws;
  auto alloc = [&](size_t bytes) {
    char* p = w;
    w += (bytes + 255) & ~(size_t)255;
    return p;
  };
  int*    bar   = (int*)alloc(256);                       // [0]=L0, [1]=L1
  float*  h0F0  = (float*)alloc(GRU_B * GRU_H * 4);
  float*  h0F1  = (float*)alloc(GRU_B * GRU_H * 4);
  bf16_t* h0B0  = (bf16_t*)alloc(GRU_B * GRU_H * 2);
  bf16_t* h0B1  = (bf16_t*)alloc(GRU_B * GRU_H * 2);
  float*  h1F0  = (float*)alloc(GRU_B * GRU_H * 4);
  float*  h1F1  = (float*)alloc(GRU_B * GRU_H * 4);
  bf16_t* h1B0  = (bf16_t*)alloc(GRU_B * GRU_H * 2);
  bf16_t* h1B1  = (bf16_t*)alloc(GRU_B * GRU_H * 2);
  bf16_t* xbf   = (bf16_t*)alloc((size_t)GRU_M * GRU_IN * 2);
  bf16_t* wih0b = (bf16_t*)alloc((size_t)GRU_3H * GRU_IN * 2);
  bf16_t* whh0b = (bf16_t*)alloc((size_t)GRU_3H * GRU_H * 2);
  bf16_t* wih1b = (bf16_t*)alloc((size_t)GRU_3H * GRU_H * 2);
  bf16_t* whh1b = (bf16_t*)alloc((size_t)GRU_3H * GRU_H * 2);
  bf16_t* out0b = (bf16_t*)alloc((size_t)GRU_M * GRU_H * 2);
  float*  xg    = (float*)alloc((size_t)GRU_M * GRU_3H * 4);

  // reset barrier + zero-init the t=0 hidden buffers (captured -> every replay)
  (void)hipMemsetAsync(bar, 0, 256, stream);
  (void)hipMemsetAsync(h0F0, 0, GRU_B * GRU_H * 4, stream);
  (void)hipMemsetAsync(h0B0, 0, GRU_B * GRU_H * 2, stream);
  (void)hipMemsetAsync(h1F0, 0, GRU_B * GRU_H * 4, stream);
  (void)hipMemsetAsync(h1B0, 0, GRU_B * GRU_H * 2, stream);

  // bf16 conversions
  {
    int n;
    n = GRU_M * GRU_IN;
    cvt_f32_bf16<<<(n + 255) / 256, 256, 0, stream>>>(x, xbf, n);
    n = GRU_3H * GRU_IN;
    cvt_f32_bf16<<<(n + 255) / 256, 256, 0, stream>>>(Wih0, wih0b, n);
    n = GRU_3H * GRU_H;
    cvt_f32_bf16<<<(n + 255) / 256, 256, 0, stream>>>(Whh0, whh0b, n);
    cvt_f32_bf16<<<(n + 255) / 256, 256, 0, stream>>>(Wih1, wih1b, n);
    cvt_f32_bf16<<<(n + 255) / 256, 256, 0, stream>>>(Whh1, whh1b, n);
  }

  (void)hipFuncSetAttribute(reinterpret_cast<const void*>(&gru_layer_kernel),
                            hipFuncAttributeMaxDynamicSharedMemorySize,
                            STEP_LDS);

  const dim3 gemm_grid(GRU_M / 128, GRU_3H / 64);

  // layer 0
  gemm_bf16_bias<<<gemm_grid, 256, 0, stream>>>(xbf, wih0b, bih0, xg, GRU_M,
                                                GRU_IN);
  gru_layer_kernel<<<STEP_WGS, 256, STEP_LDS, stream>>>(
      whh0b, bhh0, xg, h0F0, h0F1, h0B0, h0B1, out0b, bar);

  // layer 1 (xg buffer reused; out0b reused as dump for layer-1 sequence)
  gemm_bf16_bias<<<gemm_grid, 256, 0, stream>>>(out0b, wih1b, bih1, xg, GRU_M,
                                                GRU_H);
  gru_layer_kernel<<<STEP_WGS, 256, STEP_LDS, stream>>>(
      whh1b, bhh1, xg, h1F0, h1F1, h1B0, h1B1, out0b, bar + 1);

  // final projection: T even -> final h lives in buffer 0 of each layer
  out_proj<<<(GRU_B * 64) / 256, 256, 0, stream>>>(h0F0, h1F0, Wout, bout, out);
}
